// RnnModel1_1743756722578
// MI455X (gfx1250) — matrix operand order repaired and checked by reading, not compiled, not run on hardware
//
#include <hip/hip_runtime.h>
#include <hip/hip_bf16.h>

// RNN scan: h_{t+1} = tanh(x[:,t]*W_ih^T + b + h @ W_hh^T), out = h_T @ fc_w^T + fc_b
// B=1024, T=1024, H=128, C=10.
// 64 blocks x 256 threads (8 wave32s). Block owns a 16-row batch tile; wave w
// owns output-column tile n0=w (16 cols). h is exchanged between waves through
// a double-buffered f16 LDS tile with one barrier per timestep.

#define B_  1024
#define T_  1024
#define H_  128
#define C_  10

typedef __attribute__((ext_vector_type(16))) _Float16 v16h;
typedef __attribute__((ext_vector_type(8)))  _Float16 v8h;
typedef __attribute__((ext_vector_type(8)))  float    v8f;
typedef __attribute__((ext_vector_type(4)))  float    v4f;
typedef __attribute__((ext_vector_type(4)))  unsigned int v4u;

__device__ __forceinline__ float fast_tanh(float x) {
#if __has_builtin(__builtin_amdgcn_tanhf)
    return __builtin_amdgcn_tanhf(x);              // gfx1250 v_tanh_f32
#elif __has_builtin(__builtin_amdgcn_tanh_f32)
    return __builtin_amdgcn_tanh_f32(x);
#else
    float e = __builtin_amdgcn_exp2f(2.8853900817779268f * x);
    return (e - 1.0f) / (e + 1.0f);
#endif
}

// Load 16 f16 (two 16B chunks at p and p+16 halves) into a v16h A/B fragment.
__device__ __forceinline__ v16h load_frag(const _Float16* p) {
    v8h lo = *(const v8h*)p;
    v8h hi = *(const v8h*)(p + 16);
    return __builtin_shufflevector(lo, hi, 0,1,2,3,4,5,6,7,8,9,10,11,12,13,14,15);
}

__global__ __launch_bounds__(256, 1) void rnn_scan_kernel(
    const float* __restrict__ input,   // [B, T]
    const float* __restrict__ W_ih,    // [H, 1]
    const float* __restrict__ W_hh,    // [H, H] row-major [i][j]
    const float* __restrict__ b_ih,    // [H]
    const float* __restrict__ b_hh,    // [H]
    const float* __restrict__ fc_w,    // [C, H]
    const float* __restrict__ fc_b,    // [C]
    float* __restrict__ out)           // [B, C]
{
    __shared__ _Float16 lds_w[H_ * H_];      // W_hh as f16, row-major [i][j] (32 KB)
    __shared__ _Float16 h_buf[2][16 * H_];   // double-buffered h tile, f16   (2x4 KB)

    const int tid  = threadIdx.x;
    const int lane = tid & 31;            // lane within wave32
    const int n0   = tid >> 5;            // wave id == output column tile (0..7)
    const int lo   = lane & 15;
    const int hi   = lane >> 4;
    const int b0   = blockIdx.x * 16;     // batch tile base

    // ---- Stage W_hh -> LDS as f16 (cooperative, one-time) ----
    for (int idx = tid * 4; idx < H_ * H_; idx += 256 * 4) {
        v4f w = *(const v4f*)(W_hh + idx);
        lds_w[idx + 0] = (_Float16)w.x;
        lds_w[idx + 1] = (_Float16)w.y;
        lds_w[idx + 2] = (_Float16)w.z;
        lds_w[idx + 3] = (_Float16)w.w;
    }
    // ---- h_0 = 0 (both buffers) ----
    for (int idx = tid * 8; idx < 2 * 16 * H_; idx += 256 * 8) {
        v4u z = {0u, 0u, 0u, 0u};
        *(v4u*)(&h_buf[0][0] + idx) = z;
    }
    __syncthreads();

    // Per-lane input-projection constants for output column i0 = n0*16 + lo
    const int i0 = n0 * 16 + lo;
    const float wihv  = W_ih[i0];
    const float biasv = b_ih[i0] + b_hh[i0];

    // ---- This wave's 4 WMMA B-fragments of W_hh^T (register-resident).
    // B[K=j, N=i] = W_hh[i][j]; lane holds N=lo, K = 8*hi + (e&7) + 16*(e>>3) (+32*k).
    v16h bw[4];
    #pragma unroll
    for (int k = 0; k < 4; ++k)
        bw[k] = load_frag(lds_w + (n0 * 16 + lo) * H_ + k * 32 + hi * 8);

    // Row base for this lane's 8 accumulator rows: batch rows b0 + hi*8 + r
    const float* xrow = input + (size_t)(b0 + hi * 8) * T_;

    for (int t = 0; t < T_; ++t) {
        const _Float16* hsrc = &h_buf[t & 1][0];
        _Float16*       hdst = &h_buf[(t + 1) & 1][0];

        // A fragments of h_t (full K=128): lane = row M=lo
        v16h a[4];
        #pragma unroll
        for (int k = 0; k < 4; ++k)
            a[k] = load_frag(hsrc + lo * H_ + k * 32 + hi * 8);

        // Per-step scalar inputs for the 8 rows this lane accumulates
        float xr[8];
        #pragma unroll
        for (int r = 0; r < 8; ++r) xr[r] = xrow[(size_t)r * T_ + t];

        if ((t & 63) == 0) {   // uniform branch: prefetch upcoming timesteps
            #pragma unroll
            for (int r = 0; r < 8; ++r)
                __builtin_prefetch(&xrow[(size_t)r * T_ + t + 64], 0, 1);
        }

        // h @ W_hh^T for this wave's 16 columns: two independent 2-chains
        v8f acc0 = {}, acc1 = {};
        acc0 = __builtin_amdgcn_wmma_f32_16x16x32_f16(false, a[0], false, bw[0], (short)0, acc0, false, false);
        acc1 = __builtin_amdgcn_wmma_f32_16x16x32_f16(false, a[2], false, bw[2], (short)0, acc1, false, false);
        acc0 = __builtin_amdgcn_wmma_f32_16x16x32_f16(false, a[1], false, bw[1], (short)0, acc0, false, false);
        acc1 = __builtin_amdgcn_wmma_f32_16x16x32_f16(false, a[3], false, bw[3], (short)0, acc1, false, false);

        // Fused bias + input projection + tanh, write h_{t+1} slice
        #pragma unroll
        for (int r = 0; r < 8; ++r) {
            float xp = __builtin_fmaf(xr[r], wihv, biasv);
            float hn = fast_tanh(acc0[r] + acc1[r] + xp);
            // D layout: reg r, lane -> (M = r + 8*hi, N = lo); col i = n0*16 + lo
            hdst[(r + hi * 8) * H_ + n0 * 16 + lo] = (_Float16)hn;
        }

        __syncthreads();   // release stores, close cross-wave RAW/WAR for next step
    }

    // ---- Final FC: out[b0+m, c] = h_T[m,:] . fc_w[c,:] + fc_b[c]
    // h_T lives in h_buf[T_ & 1] == h_buf[0]. 160 outputs, threads 0..159.
    if (tid < 16 * C_) {
        const int m = tid & 15;
        const int c = tid >> 4;
        float s = fc_b[c];
        #pragma unroll 8
        for (int j = 0; j < H_; ++j)
            s += (float)h_buf[0][m * H_ + j] * fc_w[c * H_ + j];
        out[(size_t)(b0 + m) * C_ + c] = s;
    }
}

extern "C" void kernel_launch(void* const* d_in, const int* in_sizes, int n_in,
                              void* d_out, int out_size, void* d_ws, size_t ws_size,
                              hipStream_t stream) {
    const float* input = (const float*)d_in[0];   // [B,T,1]
    const float* W_ih  = (const float*)d_in[1];   // [H,1]
    const float* W_hh  = (const float*)d_in[2];   // [H,H]
    const float* b_ih  = (const float*)d_in[3];   // [H]
    const float* b_hh  = (const float*)d_in[4];   // [H]
    const float* fc_w  = (const float*)d_in[5];   // [C,H]
    const float* fc_b  = (const float*)d_in[6];   // [C]
    float* out = (float*)d_out;                   // [B,C]

    dim3 grid(B_ / 16);   // 64 batch tiles
    dim3 block(256);      // 8 wave32s: one per output-column tile
    rnn_scan_kernel<<<grid, block, 0, stream>>>(input, W_ih, W_hh, b_ih, b_hh,
                                                fc_w, fc_b, out);
}
